// LucidrainsWrapper_86431921864996
// MI455X (gfx1250) — matrix operand order repaired and checked by reading, not compiled, not run on hardware
//
#include <hip/hip_runtime.h>

#define NQ 4
#define KCB 1024
#define DIM 512
#define BATCH 8
#define TSEQ 2048
#define TILE_M 64              // rows (t-values) per workgroup
#define NBLK 16                // codebook entries per n-tile
#define RES_STRIDE 516         // fp32 residual row stride in LDS (16B aligned, 4-bank skew)
#define B_STRIDE 520           // f16 codebook row stride in LDS (16B aligned, 4-bank skew)
#define ALPHA_F 0.25f

typedef __attribute__((ext_vector_type(16))) _Float16 v16h;
typedef __attribute__((ext_vector_type(8)))  _Float16 v8h;
typedef __attribute__((ext_vector_type(8)))  float    v8f;

// Build A fragment (16x32 f16, M x K) for this wave's 16 rows from the fp32
// residual tile in LDS. Layout per CDNA5 ISA 7.12.2: lane = 16*hi + m,
// VGPR j holds K = (j<4?0:16) + 2*(j%4) + 8*hi, +1.
// Done ONCE per stage per k-step; result lives in VGPRs across the n-loop.
__device__ __forceinline__ v16h load_a_frag(const float* sRes, int m0w, int kk, int lane) {
  const int mr = lane & 15;
  const int hi = lane >> 4;
  const float* row = sRes + (size_t)(m0w + mr) * RES_STRIDE;
  v16h a;
#pragma unroll
  for (int j = 0; j < 8; ++j) {
    const int kidx = kk + ((j & 4) << 2) + ((j & 3) << 1) + (hi << 3);
    const float2 f = *(const float2*)(row + kidx);
    a[2 * j]     = (_Float16)f.x;
    a[2 * j + 1] = (_Float16)f.y;
  }
  return a;
}

// Build B fragment (32x16 f16, K x N): lane = 16*hi + n holds K-values
// kk + 8*hi + [0,8) and kk + 16 + 8*hi + [0,8) (matches A) -> two 16B LDS loads.
__device__ __forceinline__ v16h load_b_frag(const _Float16* sB, int kk, int lane) {
  const int e  = lane & 15;
  const int hi = lane >> 4;
  const _Float16* p = sB + e * B_STRIDE + kk + (hi << 3);
  union { v16h v; v8h h[2]; } u;
  u.h[0] = *(const v8h*)(p);
  u.h[1] = *(const v8h*)(p + 16);
  return u.v;
}

__global__ void __launch_bounds__(128, 2)
rvq_main(const float* __restrict__ z_e, const float* __restrict__ cbF32,
         const _Float16* __restrict__ cbF16, const float* __restrict__ c2,
         float* __restrict__ out) {
  extern __shared__ __align__(16) char smem[];
  float*    sRes = (float*)smem;                                        // [TILE_M][RES_STRIDE]
  _Float16* sB   = (_Float16*)(smem + (size_t)TILE_M * RES_STRIDE * 4); // [NBLK][B_STRIDE]
  int*      sIdx = (int*)((char*)sB + (size_t)NBLK * B_STRIDE * 2);     // [TILE_M]
  float*    sRed = (float*)(sIdx + TILE_M);                             // [4]

  const int tid  = threadIdx.x;
  const int lane = tid & 31;
  const int w    = tid >> 5;       // wave 0..3 -> M-subtile
  const int m0w  = w << 4;
  const int hi   = lane >> 4;
  const int col  = lane & 15;

  const int tile = blockIdx.x;                 // 256 tiles: 32 per batch entry
  const int b    = tile >> 5;
  const int t0   = (tile & 31) * TILE_M;
  const size_t baseBT = (size_t)b * DIM * TSEQ + (size_t)t0;

  // ---- load residual tile (transpose [d][t] -> LDS [t_local][d]), coalesced on t
  for (int it = 0; it < DIM / 2; ++it) {
    const int d = (it << 1) + (tid >> 6);
    const int m = tid & 63;
    sRes[m * RES_STRIDE + d] = z_e[baseBT + (size_t)d * TSEQ + m];
  }
  __syncthreads();

  // ---- all 4 quantizer stages on the resident tile
  for (int q = 0; q < NQ; ++q) {
    const _Float16* cbq16 = cbF16 + (size_t)q * KCB * DIM;
    const float*    c2q   = c2 + q * KCB;

    // A fragments for this wave's 16 rows, full K=512: 16 frags = 128 VGPRs,
    // register-resident for the whole codebook scan (no cvts in the hot loop).
    v16h afrag[16];
#pragma unroll
    for (int ks = 0; ks < 16; ++ks) afrag[ks] = load_a_frag(sRes, m0w, ks * 32, lane);

    float best[8];
    int   bidx[8];
#pragma unroll
    for (int vi = 0; vi < 8; ++vi) { best[vi] = 3.4e38f; bidx[vi] = 0; }

    for (int nt = 0; nt < KCB / NBLK; ++nt) {
      const int n0 = nt * NBLK;
      __syncthreads();
      {  // cooperative stage of 16 codebook entries (f16) into LDS, shared by all waves
        const int e    = tid >> 3;
        const int part = tid & 7;
        const uint4* src = (const uint4*)(cbq16 + (size_t)(n0 + e) * DIM + part * 64);
        uint4* dst = (uint4*)(sB + e * B_STRIDE + part * 64);
        dst[0] = src[0]; dst[1] = src[1]; dst[2] = src[2]; dst[3] = src[3];
        if (nt + 1 < KCB / NBLK)  // hint next tile toward L2/L0
          __builtin_prefetch(cbq16 + (size_t)(n0 + NBLK + e) * DIM + part * 64, 0, 0);
      }
      __syncthreads();

      // dots: 16 k-steps of V_WMMA_F32_16X16X32_F16, two independent chains;
      // steady state per WMMA: 2x ds_load_b128 (B) + the WMMA itself.
      v8f c0 = {}; v8f c1 = {};
#pragma unroll
      for (int ks = 0; ks < 16; ks += 2) {
        v16h b0 = load_b_frag(sB, ks * 32, lane);
        c0 = __builtin_amdgcn_wmma_f32_16x16x32_f16(false, afrag[ks], false, b0,
                                                    (short)0, c0, false, false);
        v16h b1 = load_b_frag(sB, (ks + 1) * 32, lane);
        c1 = __builtin_amdgcn_wmma_f32_16x16x32_f16(false, afrag[ks + 1], false, b1,
                                                    (short)0, c1, false, false);
      }

      const float c2n = c2q[n0 + col];
      const int   n   = n0 + col;
#pragma unroll
      for (int vi = 0; vi < 8; ++vi) {  // lane holds rows vi + 8*hi, column n
        const float dist = c2n - 2.0f * (c0[vi] + c1[vi]);
        if (dist < best[vi]) { best[vi] = dist; bidx[vi] = n; }
      }
    }

    // cross-lane argmin over the 16 columns in each half (C layout preserved)
#pragma unroll
    for (int mask = 1; mask <= 8; mask <<= 1) {
#pragma unroll
      for (int vi = 0; vi < 8; ++vi) {
        const float ob = __shfl_xor(best[vi], mask, 32);
        const int   oi = __shfl_xor(bidx[vi], mask, 32);
        if (ob < best[vi] || (ob == best[vi] && oi < bidx[vi])) {
          best[vi] = ob; bidx[vi] = oi;
        }
      }
    }
    if (col == 0) {
#pragma unroll
      for (int vi = 0; vi < 8; ++vi) sIdx[m0w + (hi << 3) + vi] = bidx[vi];
    }
    __syncthreads();

    // gather fp32 codeword + exact residual update in LDS (wave owns its rows)
    const float* cbq32 = cbF32 + (size_t)q * KCB * DIM;
    for (int r = 0; r < 16; ++r) {
      const int row = m0w + r;
      const int ci  = sIdx[row];
      const float4* crow = (const float4*)(cbq32 + (size_t)ci * DIM);
      float4* rrow = (float4*)(sRes + (size_t)row * RES_STRIDE);
      for (int j = lane; j < DIM / 4; j += 32) {
        float4 cv = crow[j];
        float4 rv = rrow[j];
        rv.x -= cv.x; rv.y -= cv.y; rv.z -= cv.z; rv.w -= cv.w;
        rrow[j] = rv;
      }
    }
    __syncthreads();
  }

  // ---- z_q = z_e - r_final  (Σ quant identity); loss_last = mean(r_final^2)
  float ssq = 0.f;
  for (int it = 0; it < DIM / 2; ++it) {
    const int d = (it << 1) + (tid >> 6);
    const int m = tid & 63;
    const float r = sRes[m * RES_STRIDE + d];
    ssq += r * r;
    const size_t gi = baseBT + (size_t)d * TSEQ + m;
    out[gi] = z_e[gi] - r;
  }
#pragma unroll
  for (int mask = 16; mask >= 1; mask >>= 1) ssq += __shfl_xor(ssq, mask, 32);
  if (lane == 0) sRed[w] = ssq;
  __syncthreads();
  if (tid == 0) {
    const float tot = sRed[0] + sRed[1] + sRed[2] + sRed[3];
    atomicAdd(out + (size_t)BATCH * DIM * TSEQ,
              tot * (ALPHA_F / (float)((size_t)BATCH * DIM * TSEQ)));
  }
}

// One block per codebook entry: f16 copy + ||c||^2
__global__ void rvq_prep(const float* __restrict__ cb, _Float16* __restrict__ cbF16,
                         float* __restrict__ c2) {
  const int e = blockIdx.x;  // 0 .. NQ*KCB-1
  const float* row = cb + (size_t)e * DIM;
  _Float16* orow = cbF16 + (size_t)e * DIM;
  float s = 0.f;
  for (int j = threadIdx.x; j < DIM; j += 128) {
    const float v = row[j];
    s += v * v;
    orow[j] = (_Float16)v;
  }
#pragma unroll
  for (int mask = 16; mask >= 1; mask >>= 1) s += __shfl_xor(s, mask, 32);
  __shared__ float red[4];
  if ((threadIdx.x & 31) == 0) red[threadIdx.x >> 5] = s;
  __syncthreads();
  if (threadIdx.x == 0) c2[e] = red[0] + red[1] + red[2] + red[3];
}

__global__ void rvq_init_tail(float* __restrict__ out) {
  if (threadIdx.x < 2) out[(size_t)BATCH * DIM * TSEQ + threadIdx.x] = 0.f;
}

extern "C" void kernel_launch(void* const* d_in, const int* in_sizes, int n_in,
                              void* d_out, int out_size, void* d_ws, size_t ws_size,
                              hipStream_t stream) {
  const float* z_e = (const float*)d_in[0];   // [8, 512, 2048]
  const float* cb  = (const float*)d_in[1];   // [4, 1024, 512]
  float* out = (float*)d_out;                 // z_q flat + loss + perplexity

  _Float16* cbF16 = (_Float16*)d_ws;          // 4 MB
  float* c2 = (float*)((char*)d_ws + (size_t)NQ * KCB * DIM * sizeof(_Float16)); // 16 KB

  rvq_init_tail<<<1, 32, 0, stream>>>(out);
  rvq_prep<<<NQ * KCB, 128, 0, stream>>>(cb, cbF16, c2);

  const size_t smem = (size_t)TILE_M * RES_STRIDE * sizeof(float)
                    + (size_t)NBLK * B_STRIDE * sizeof(_Float16)
                    + (size_t)TILE_M * sizeof(int) + 16 * sizeof(float);
  rvq_main<<<(BATCH * TSEQ) / TILE_M, 128, smem, stream>>>(z_e, cb, cbF16, c2, out);
}
